// MultiHeadAttention_70875550318826
// MI455X (gfx1250) — matrix operand order repaired and checked by reading, not compiled, not run on hardware
//
#include <hip/hip_runtime.h>
#include <hip/hip_bf16.h>
#include <math.h>

// ---------------------------------------------------------------------------
// MI455X (gfx1250) multi-head causal attention, bf16 WMMA pipeline.
// B=4, S=2048, D=1024, H=16, HD=64.
//
// All matmuls use v_wmma_f32_16x16x32_bf16 (K=32/issue, f32 accum).
// Register-blocked so each fragment load feeds multiple WMMAs.
// Softmax + accumulation entirely in fp32. EXEC all-ones around every WMMA.
// ---------------------------------------------------------------------------

constexpr int B_  = 4;
constexpr int S_  = 2048;
constexpr int D_  = 1024;
constexpr int H_  = 16;
constexpr int HD_ = 64;
constexpr float SCALE_ = 0.125f;   // 1/sqrt(64)

typedef __bf16 bf16;
typedef __attribute__((ext_vector_type(16))) __bf16 v16bf;
typedef __attribute__((ext_vector_type(8)))  __bf16 v8bf;
typedef __attribute__((ext_vector_type(8)))  float  v8f;

// fp32 -> bf16, round-to-nearest-even
__device__ __forceinline__ bf16 f2bf(float f) {
  unsigned u = __builtin_bit_cast(unsigned, f);
  u += 0x7FFFu + ((u >> 16) & 1u);
  unsigned short h = (unsigned short)(u >> 16);
  return __builtin_bit_cast(bf16, h);
}

// Two 16B loads -> one 32B WMMA fragment register block.
__device__ __forceinline__ v16bf load16(const bf16* p0, const bf16* p1) {
  v8bf a = *(const v8bf*)p0;
  v8bf b = *(const v8bf*)p1;
  v16bf r;
#pragma unroll
  for (int i = 0; i < 8; ++i) { r[i] = a[i]; r[i + 8] = b[i]; }
  return r;
}

__device__ __forceinline__ v8f wmma_bf16(v16bf a, v16bf b, v8f c) {
  return __builtin_amdgcn_wmma_f32_16x16x32_bf16(
      false, a, false, b, (short)0, c, false, false);
}

// ---------------------------------------------------------------------------
// 1) fp32 -> bf16 conversion
// ---------------------------------------------------------------------------
__global__ void cvt_f32_bf16(const float* __restrict__ in,
                             bf16* __restrict__ out, int n) {
  int i = blockIdx.x * blockDim.x + threadIdx.x;
  if (i < n) out[i] = f2bf(in[i]);
}

// ---------------------------------------------------------------------------
// 2) Pre-swizzle Wq/Wk/Wv into WMMA B-fragment layout.
//    Fragment layout: [z=w*H+h][ntile(4)][kstep(32)][lane(32)][16 bf16].
//    B mapping: lane -> col n = ntile*16 + (lane&15);
//               element e -> k = kstep*32 + 16*(lane>>4) + e.
//    grid = (32, 4, 48), block = 32.
// ---------------------------------------------------------------------------
__global__ __launch_bounds__(32) void swz_qkv(const float* __restrict__ Wq,
                                              const float* __restrict__ Wk,
                                              const float* __restrict__ Wv,
                                              bf16* __restrict__ frag) {
  const int lane = threadIdx.x;
  const int hi = lane >> 4, lo = lane & 15;
  const int ks = blockIdx.x, nt = blockIdx.y, z = blockIdx.z;
  const int w = z / H_, h = z % H_;
  const float* W = (w == 0 ? Wq : (w == 1 ? Wk : Wv)) + (size_t)h * D_ * HD_;
  const int n = nt * 16 + lo;
  const int kb = ks * 32 + 16 * hi;
  v8bf a, b;
#pragma unroll
  for (int e = 0; e < 8; ++e) {
    a[e] = f2bf(W[(size_t)(kb + e) * HD_ + n]);
    b[e] = f2bf(W[(size_t)(kb + 8 + e) * HD_ + n]);
  }
  bf16* dst = frag + ((((size_t)z * 4 + nt) * 32 + ks) * 32 + lane) * 16;
  *(v8bf*)dst = a;
  *(v8bf*)(dst + 8) = b;
}

// ---------------------------------------------------------------------------
// 3) QKV projections: out[b,h,s,e] = sum_d x[b,s,d] * W[h,d,e].
//    Register-blocked: one wave -> 32x64 output (2 m-tiles x 4 e-tiles),
//    8 WMMAs per K-step off 4 A-loads + 8 B-loads.
//    grid = (256, 1, 48 = w*H+h), block = 32.
//    Q,K stored [b,h,s,e] (e contiguous); V stored transposed [b,h,e,t].
// ---------------------------------------------------------------------------
__global__ __launch_bounds__(32) void qkv_gemm(const bf16* __restrict__ xb,
                                               const bf16* __restrict__ wfrag,
                                               bf16* __restrict__ Q,
                                               bf16* __restrict__ K,
                                               bf16* __restrict__ Vt) {
  const int lane = threadIdx.x;
  const int hi = lane >> 4, lo = lane & 15;
  const int mt = blockIdx.x, z = blockIdx.z;
  const int w = z / H_, h = z % H_;

  const bf16* arow0 = xb + (size_t)(mt * 32 + lo) * D_;
  const bf16* arow1 = arow0 + (size_t)16 * D_;
  const bf16* bbase = wfrag + (size_t)z * 4 * 16384 + (size_t)lane * 16;

  v8f c[2][4] = {};
#pragma unroll 2
  for (int ks = 0; ks < 32; ++ks) {
    const int kb = ks * 32 + 16 * hi;
    v16bf a0 = load16(arow0 + kb, arow0 + kb + 8);
    v16bf a1 = load16(arow1 + kb, arow1 + kb + 8);
#pragma unroll
    for (int nt = 0; nt < 4; ++nt) {
      const bf16* bp = bbase + (size_t)nt * 16384 + (size_t)ks * 512;
      v16bf b = load16(bp, bp + 8);
      c[0][nt] = wmma_bf16(a0, b, c[0][nt]);
      c[1][nt] = wmma_bf16(a1, b, c[1][nt]);
    }
  }

#pragma unroll
  for (int mi = 0; mi < 2; ++mi) {
#pragma unroll
    for (int nt = 0; nt < 4; ++nt) {
      const int e = nt * 16 + lo;
#pragma unroll
      for (int r = 0; r < 8; ++r) {
        const int m = mt * 32 + mi * 16 + r + 8 * hi;
        const int bb = m / S_, s = m % S_;
        const bf16 val = f2bf(c[mi][nt][r]);
        if (w == 0)
          Q[((size_t)(bb * H_ + h) * S_ + s) * HD_ + e] = val;
        else if (w == 1)
          K[((size_t)(bb * H_ + h) * S_ + s) * HD_ + e] = val;
        else
          Vt[((size_t)(bb * H_ + h) * HD_ + e) * S_ + s] = val;
      }
    }
  }
}

// ---------------------------------------------------------------------------
// 4) Flash-style causal attention. One wave per (b*H+h, 16-row q tile).
//    Key steps are 64 wide: 8 score WMMAs + 8 PV WMMAs per step; the softmax
//    shuffle butterflies and LDS transpose are amortized over 64 keys.
//    grid = (128, 64), block = 32. 2 KB LDS for the P transpose.
// ---------------------------------------------------------------------------
__global__ __launch_bounds__(32) void attn_kernel(const bf16* __restrict__ Q,
                                                  const bf16* __restrict__ Km,
                                                  const bf16* __restrict__ Vt,
                                                  bf16* __restrict__ concat) {
  __shared__ __align__(16) bf16 pb[16 * 64];

  const int lane = threadIdx.x;
  const int hi = lane >> 4, lo = lane & 15;
  const int qt = blockIdx.x;           // 0..127
  const int bh = blockIdx.y;           // b*H + h
  const int b = bh / H_, h = bh % H_;

  const bf16* q_ = Q  + (size_t)bh * S_ * HD_;
  const bf16* k_ = Km + (size_t)bh * S_ * HD_;
  const bf16* v_ = Vt + (size_t)bh * HD_ * S_;

  // Q A-fragments (row = qt*16+lo), two 32-wide e chunks.
  const bf16* qrow = q_ + (size_t)(qt * 16 + lo) * HD_;
  v16bf qa[2];
#pragma unroll
  for (int eh = 0; eh < 2; ++eh)
    qa[eh] = load16(qrow + eh * 32 + 16 * hi, qrow + eh * 32 + 16 * hi + 8);

  v8f acc[4] = {};
  float mrow[8], lrow[8];
#pragma unroll
  for (int r = 0; r < 8; ++r) { mrow[r] = -INFINITY; lrow[r] = 0.f; }

  const int nsteps = ((qt * 16 + 15) >> 6) + 1;
  for (int ksp = 0; ksp < nsteps; ++ksp) {
    const int t0 = ksp * 64;

    // ---- scores: four 16-col C fragments over 64 keys ----
    v8f sc[4] = {};
#pragma unroll
    for (int cf = 0; cf < 4; ++cf) {
      const bf16* krow = k_ + (size_t)(t0 + cf * 16 + lo) * HD_;
#pragma unroll
      for (int eh = 0; eh < 2; ++eh) {
        v16bf kf = load16(krow + eh * 32 + 16 * hi,
                          krow + eh * 32 + 16 * hi + 8);
        sc[cf] = wmma_bf16(qa[eh], kf, sc[cf]);
      }
    }

    // ---- causal mask + online softmax (rows live in 16-lane halves) ----
    float alpha[8], pv[4][8];
#pragma unroll
    for (int r = 0; r < 8; ++r) {
      const int sg = qt * 16 + r + 8 * hi;
      float sv[4];
      float tm = -INFINITY;
#pragma unroll
      for (int cf = 0; cf < 4; ++cf) {
        const int tg = t0 + cf * 16 + lo;
        sv[cf] = (tg <= sg) ? sc[cf][r] * SCALE_ : -INFINITY;
        tm = fmaxf(tm, sv[cf]);
      }
#pragma unroll
      for (int mk = 1; mk < 16; mk <<= 1)
        tm = fmaxf(tm, __shfl_xor(tm, mk, 32));
      const float mn = fmaxf(mrow[r], tm);
      float ps = 0.f;
#pragma unroll
      for (int cf = 0; cf < 4; ++cf) {
        const float e = __expf(sv[cf] - mn);
        pv[cf][r] = e;
        ps += e;
      }
#pragma unroll
      for (int mk = 1; mk < 16; mk <<= 1)
        ps += __shfl_xor(ps, mk, 32);
      alpha[r] = __expf(mrow[r] - mn);
      mrow[r] = mn;
      lrow[r] = lrow[r] * alpha[r] + ps;
    }

    // ---- transpose P (C layout -> A fragments) through LDS ----
#pragma unroll
    for (int cf = 0; cf < 4; ++cf)
#pragma unroll
      for (int r = 0; r < 8; ++r)
        pb[(r + 8 * hi) * 64 + cf * 16 + lo] = f2bf(pv[cf][r]);
    asm volatile("s_wait_dscnt 0x0" ::: "memory");
    __builtin_amdgcn_wave_barrier();
    const bf16* prow = pb + (size_t)lo * 64;
    v16bf pa[2];
#pragma unroll
    for (int th = 0; th < 2; ++th)
      pa[th] = load16(prow + th * 32 + 16 * hi, prow + th * 32 + 16 * hi + 8);
    __builtin_amdgcn_wave_barrier();

    // ---- acc = acc*alpha + P @ V  (V transposed: [e][t], t contiguous) ----
#pragma unroll
    for (int nt = 0; nt < 4; ++nt) {
      v8f ac = acc[nt];
#pragma unroll
      for (int r = 0; r < 8; ++r) ac[r] *= alpha[r];
      const bf16* vrow = v_ + (size_t)(nt * 16 + lo) * S_ + t0;
#pragma unroll
      for (int th = 0; th < 2; ++th) {
        v16bf vb = load16(vrow + th * 32 + 16 * hi,
                          vrow + th * 32 + 16 * hi + 8);
        ac = wmma_bf16(pa[th], vb, ac);
      }
      acc[nt] = ac;
    }
  }

  // ---- normalize and write concat [b,s, h*64+e] as bf16 ----
#pragma unroll
  for (int nt = 0; nt < 4; ++nt) {
    const int e = h * HD_ + nt * 16 + lo;
#pragma unroll
    for (int r = 0; r < 8; ++r) {
      const int s = qt * 16 + r + 8 * hi;
      concat[(size_t)(b * S_ + s) * D_ + e] = f2bf(acc[nt][r] / lrow[r]);
    }
  }
}

// ---------------------------------------------------------------------------
// 5) Output projection: out[m,n] = sum_k concat[m,k] * W_O[n,k].
//    W_O row-major [n][k] means the B-fragment k-run is contiguous: no swizzle.
//    Register-blocked: one wave -> 64x32 output (4 m x 2 n), 8 WMMAs/K-step.
//    grid = (128, 32), block = 32.
// ---------------------------------------------------------------------------
__global__ __launch_bounds__(32) void oproj_gemm(const bf16* __restrict__ concat,
                                                 const bf16* __restrict__ wo,
                                                 float* __restrict__ out) {
  const int lane = threadIdx.x;
  const int hi = lane >> 4, lo = lane & 15;
  const int mt = blockIdx.x, nt = blockIdx.y;

  const bf16* arow[4];
#pragma unroll
  for (int mi = 0; mi < 4; ++mi)
    arow[mi] = concat + (size_t)(mt * 64 + mi * 16 + lo) * D_;
  const bf16* brow[2];
#pragma unroll
  for (int ni = 0; ni < 2; ++ni)
    brow[ni] = wo + (size_t)(nt * 32 + ni * 16 + lo) * D_;

  v8f c[4][2] = {};
#pragma unroll 2
  for (int ks = 0; ks < 32; ++ks) {
    const int kb = ks * 32;
    v16bf a[4], bfrag[2];
#pragma unroll
    for (int mi = 0; mi < 4; ++mi)
      a[mi] = load16(arow[mi] + kb + 16 * hi, arow[mi] + kb + 16 * hi + 8);
#pragma unroll
    for (int ni = 0; ni < 2; ++ni)
      bfrag[ni] = load16(brow[ni] + kb + 16 * hi, brow[ni] + kb + 16 * hi + 8);
#pragma unroll
    for (int mi = 0; mi < 4; ++mi)
#pragma unroll
      for (int ni = 0; ni < 2; ++ni)
        c[mi][ni] = wmma_bf16(a[mi], bfrag[ni], c[mi][ni]);
  }

#pragma unroll
  for (int mi = 0; mi < 4; ++mi) {
#pragma unroll
    for (int ni = 0; ni < 2; ++ni) {
      const int n = nt * 32 + ni * 16 + lo;
#pragma unroll
      for (int r = 0; r < 8; ++r) {
        const int m = mt * 64 + mi * 16 + r + 8 * hi;
        out[(size_t)m * D_ + n] = c[mi][ni][r];
      }
    }
  }
}

// ---------------------------------------------------------------------------
// Launch
// ---------------------------------------------------------------------------
extern "C" void kernel_launch(void* const* d_in, const int* in_sizes, int n_in,
                              void* d_out, int out_size, void* d_ws, size_t ws_size,
                              hipStream_t stream) {
  (void)in_sizes; (void)n_in; (void)out_size; (void)ws_size;

  const float* x   = (const float*)d_in[0];
  const float* Wq  = (const float*)d_in[1];
  const float* Wk  = (const float*)d_in[2];
  const float* Wv  = (const float*)d_in[3];
  const float* W_O = (const float*)d_in[4];

  // Workspace layout (bytes); total = 88 MiB.
  char* ws = (char*)d_ws;
  const size_t NX  = (size_t)B_ * S_ * D_;        // 8,388,608
  const size_t NWO = (size_t)D_ * D_;             // 1,048,576
  const size_t NQ  = (size_t)B_ * H_ * S_ * HD_;  // 8,388,608
  bf16* xb     = (bf16*)(ws);                       // 16 MiB
  bf16* wfrag  = (bf16*)(ws + (NX) * 2);            //  6 MiB
  bf16* wob    = (bf16*)(ws + (NX + 3145728) * 2);  //  2 MiB
  bf16* Q      = (bf16*)(ws + (NX + 3145728 + NWO) * 2);
  bf16* K      = Q + NQ;
  bf16* Vt     = K + NQ;
  bf16* concat = Vt + NQ;

  cvt_f32_bf16<<<(unsigned)((NX + 255) / 256), 256, 0, stream>>>(x, xb, (int)NX);
  cvt_f32_bf16<<<(unsigned)((NWO + 255) / 256), 256, 0, stream>>>(W_O, wob, (int)NWO);
  swz_qkv<<<dim3(32, 4, 3 * H_), 32, 0, stream>>>(Wq, Wk, Wv, wfrag);
  qkv_gemm<<<dim3(B_ * S_ / 32, 1, 3 * H_), 32, 0, stream>>>(xb, wfrag, Q, K, Vt);
  attn_kernel<<<dim3(S_ / 16, B_ * H_), 32, 0, stream>>>(Q, K, Vt, concat);
  oproj_gemm<<<dim3(B_ * S_ / 64, D_ / 32), 32, 0, stream>>>(concat, wob, (float*)d_out);
}
